// GroupedQueryAttention_57887569216001
// MI455X (gfx1250) — matrix operand order repaired and checked by reading, not compiled, not run on hardware
//
#include <hip/hip_runtime.h>
#include <hip/hip_bf16.h>
#include <math.h>

typedef __bf16 bf16_t;
typedef __attribute__((ext_vector_type(16))) __bf16 v16bf;
typedef __attribute__((ext_vector_type(8)))  float  v8f;
typedef __attribute__((ext_vector_type(4)))  unsigned int uint4v;

#define D_MODEL 1024
#define NHEAD   16
#define NUM_KV  4
#define D_K     64
#define B_      2
#define S_      2048
#define MROWS   (B_ * S_)        /* 4096 */
#define KV_W    (NUM_KV * D_K)   /* 256  */

// ---------------------------------------------------------------------------
// WMMA fragment helpers (wave32, 16x16x32 bf16, contiguous k order, same in A and B)
// ---------------------------------------------------------------------------
__device__ __forceinline__ v8f wmma_bf16(v16bf a, v16bf b, v8f c) {
  return __builtin_amdgcn_wmma_f32_16x16x32_bf16(
      /*neg_a=*/false, a, /*neg_b=*/false, b,
      /*c_mod=*/(short)0, c, /*reuse_a=*/false, /*reuse_b=*/false);
}

// A (16x32, MxK) from row-major LDS tile. row = lane&15;
// lanes 0-15 hold K = 0..15, lanes 16-31 hold K = 16..31 (contiguous).
__device__ __forceinline__ v16bf load_a_frag(const bf16_t* base, int ld) {
  int lane = threadIdx.x & 31;
  const bf16_t* p = base + (lane & 15) * ld + (lane >> 4) * 16;
  v16bf a;
#pragma unroll
  for (int j = 0; j < 8; ++j)  a[j] = p[j];
#pragma unroll
  for (int j = 8; j < 16; ++j) a[j] = p[j];
  return a;
}

// B (32x16, KxN) where the source tile is row-major [N][K]  (i.e. B = srcT).
// N = lane&15; lanes 0-15 hold K=0..15, lanes 16-31 hold K=16..31 (contiguous).
__device__ __forceinline__ v16bf load_bT_frag(const bf16_t* base, int ld) {
  int lane = threadIdx.x & 31;
  const bf16_t* p = base + (lane & 15) * ld + (lane >> 4) * 16;
  v16bf b;
#pragma unroll
  for (int j = 0; j < 16; ++j) b[j] = p[j];
  return b;
}

// B (32x16, KxN) where the source tile is row-major [K][N] (direct, strided).
__device__ __forceinline__ v16bf load_b_frag(const bf16_t* base, int ld) {
  int lane = threadIdx.x & 31;
  const bf16_t* p = base + (lane >> 4) * 16 * ld + (lane & 15);
  v16bf b;
#pragma unroll
  for (int j = 0; j < 16; ++j) b[j] = p[j * ld];
  return b;
}

// ---------------------------------------------------------------------------
// fp32 -> bf16 conversion
// ---------------------------------------------------------------------------
__global__ void cvt_f32_bf16(const float* __restrict__ x, bf16_t* __restrict__ y, int n) {
  int i = blockIdx.x * 256 + threadIdx.x;
  if (i < n) y[i] = (bf16_t)x[i];
}

// ---------------------------------------------------------------------------
// Generic bf16 GEMM:  C[M,N] = A[M,K] @ W[N,K]^T + bias[N]
// 128x128 block tile, 256 threads (8 waves), each wave owns 32x64.
// Register-staged double-buffered LDS: one barrier per 32-wide K chunk.
// ---------------------------------------------------------------------------
#define GT_LD 40   /* 32 + 8 pad, keeps rows 16B aligned */
__global__ void __launch_bounds__(256)
gemm_bf16_kernel(const bf16_t* __restrict__ A, const bf16_t* __restrict__ W,
                 const float* __restrict__ bias,
                 bf16_t* __restrict__ Cb, float* __restrict__ Cf,
                 int M, int N, int K)
{
  __shared__ bf16_t As[2][128][GT_LD];
  __shared__ bf16_t Ws[2][128][GT_LD];

  const int tid  = threadIdx.x;
  const int wave = tid >> 5;
  const int lane = tid & 31;
  const int half = lane >> 4;
  const int nc   = lane & 15;
  const int wm   = wave & 3;        // 32-row subtile
  const int wn   = wave >> 2;       // 64-col subtile
  const int m0   = blockIdx.y * 128;
  const int n0   = blockIdx.x * 128;

  // staging: 128 rows x 4 16B-chunks per tile; this thread covers rows r0, r0+64
  const int r0 = tid >> 2;
  const int c0 = (tid & 3) << 3;

  // tile 0 -> LDS buffer 0
  {
    uint4v a0 = *(const uint4v*)&A[(size_t)(m0 + r0) * K + c0];
    uint4v a1 = *(const uint4v*)&A[(size_t)(m0 + r0 + 64) * K + c0];
    uint4v w0 = *(const uint4v*)&W[(size_t)(n0 + r0) * K + c0];
    uint4v w1 = *(const uint4v*)&W[(size_t)(n0 + r0 + 64) * K + c0];
    *(uint4v*)&As[0][r0][c0]      = a0;
    *(uint4v*)&As[0][r0 + 64][c0] = a1;
    *(uint4v*)&Ws[0][r0][c0]      = w0;
    *(uint4v*)&Ws[0][r0 + 64][c0] = w1;
  }
  __syncthreads();

  const v8f vzero = {0.f, 0.f, 0.f, 0.f, 0.f, 0.f, 0.f, 0.f};
  v8f acc[2][4];
#pragma unroll
  for (int mt = 0; mt < 2; ++mt)
#pragma unroll
    for (int nt = 0; nt < 4; ++nt) acc[mt][nt] = vzero;

  const int nchunk = K >> 5;
  for (int i = 0; i < nchunk; ++i) {
    const int buf = i & 1;
    const bool hasNext = (i + 1) < nchunk;
    uint4v na0, na1, nw0, nw1;
    if (hasNext) {
      int k0 = (i + 1) << 5;
      na0 = *(const uint4v*)&A[(size_t)(m0 + r0) * K + k0 + c0];
      na1 = *(const uint4v*)&A[(size_t)(m0 + r0 + 64) * K + k0 + c0];
      nw0 = *(const uint4v*)&W[(size_t)(n0 + r0) * K + k0 + c0];
      nw1 = *(const uint4v*)&W[(size_t)(n0 + r0 + 64) * K + k0 + c0];
    }

    // 8 WMMAs on current buffer (2 A-frags reused across 4 B-frags)
    v16bf af0 = load_a_frag(&As[buf][wm * 32][0], GT_LD);
    v16bf af1 = load_a_frag(&As[buf][wm * 32 + 16][0], GT_LD);
#pragma unroll
    for (int nt = 0; nt < 4; ++nt) {
      v16bf bf = load_bT_frag(&Ws[buf][wn * 64 + nt * 16][0], GT_LD);
      acc[0][nt] = wmma_bf16(af0, bf, acc[0][nt]);
      acc[1][nt] = wmma_bf16(af1, bf, acc[1][nt]);
    }

    if (hasNext) {
      const int nb = buf ^ 1;
      *(uint4v*)&As[nb][r0][c0]      = na0;
      *(uint4v*)&As[nb][r0 + 64][c0] = na1;
      *(uint4v*)&Ws[nb][r0][c0]      = nw0;
      *(uint4v*)&Ws[nb][r0 + 64][c0] = nw1;
    }
    __syncthreads();
  }

#pragma unroll
  for (int mt = 0; mt < 2; ++mt) {
#pragma unroll
    for (int nt = 0; nt < 4; ++nt) {
#pragma unroll
      for (int r = 0; r < 8; ++r) {
        int gm = m0 + wm * 32 + mt * 16 + r + half * 8;
        int gn = n0 + wn * 64 + nt * 16 + nc;
        float v = acc[mt][nt][r] + bias[gn];
        if (Cf) Cf[(size_t)gm * N + gn] = v;
        else    Cb[(size_t)gm * N + gn] = (bf16_t)v;
      }
    }
  }
}

// ---------------------------------------------------------------------------
// In-place RoPE on [rows, nheads*64] bf16; pos = row % seqlen.
// ---------------------------------------------------------------------------
__global__ void rope_kernel(bf16_t* __restrict__ X, int rows, int nheads, int seqlen) {
  int idx = blockIdx.x * 256 + threadIdx.x;
  int total = rows * nheads * 32;
  if (idx >= total) return;
  int i   = idx & 31;
  int h   = (idx >> 5) % nheads;
  int row = idx / (32 * nheads);
  int pos = row % seqlen;
  float inv_freq = __expf(-(float)i * (9.210340371976184f / 32.0f)); // 10000^(-i/32)
  float theta = (float)pos * inv_freq;
  float s, c;
  sincosf(theta, &s, &c);
  bf16_t* p = X + (size_t)row * (nheads * 64) + h * 64 + i;
  float x1 = (float)p[0];
  float x2 = (float)p[32];
  p[0]  = (bf16_t)(x1 * c - x2 * s);
  p[32] = (bf16_t)(x2 * c + x1 * s);
}

// ---------------------------------------------------------------------------
// Flash attention: 128-row Q block of one (b,h) per 256-thread block (8 waves,
// 16 rows each). Online softmax over 32 key tiles of 64. GQA via kv = h>>2.
// Next K/V tile is prefetched (global_prefetch_b8) while current one computes.
// ---------------------------------------------------------------------------
#define AT_LD 72   /* 64 + 8 pad, rows 16B aligned */
__global__ void __launch_bounds__(256)
attn_kernel(const bf16_t* __restrict__ Q, const bf16_t* __restrict__ Kl,
            const bf16_t* __restrict__ Vl, bf16_t* __restrict__ Ctx)
{
  __shared__ bf16_t Qs[128][AT_LD];
  __shared__ bf16_t Ks[64][AT_LD];
  __shared__ bf16_t Vs[64][AT_LD];
  __shared__ bf16_t Ps[128][AT_LD];
  __shared__ float  Sf[128][65];
  __shared__ float  m_s[128], l_s[128], al_s[128];

  const int tid  = threadIdx.x;
  const int wave = tid >> 5;       // 0..7
  const int lane = tid & 31;
  const int half = lane >> 4;
  const int nc   = lane & 15;
  const int qb   = blockIdx.x;     // 0..15
  const int h    = blockIdx.y;     // 0..15
  const int b    = blockIdx.z;     // 0..1
  const int kv   = h >> 2;         // shared KV head
  const int s0   = qb * 128;
  const float scale = 0.125f;      // 1/sqrt(64)

  // Q tile (128 x 64 bf16)
  for (int idx = tid; idx < 128 * 8; idx += 256) {
    int r = idx >> 3, c = (idx & 7) << 3;
    *(uint4v*)&Qs[r][c] =
        *(const uint4v*)&Q[(size_t)(b * S_ + s0 + r) * D_MODEL + h * D_K + c];
  }
  if (tid < 128) { m_s[tid] = -3.0e38f; l_s[tid] = 0.f; }

  const v8f vzero = {0.f, 0.f, 0.f, 0.f, 0.f, 0.f, 0.f, 0.f};
  v8f acc[4];
#pragma unroll
  for (int t = 0; t < 4; ++t) acc[t] = vzero;

  for (int kt = 0; kt < S_ / 64; ++kt) {
    __syncthreads();
    for (int idx = tid; idx < 64 * 8; idx += 256) {
      int r = idx >> 3, c = (idx & 7) << 3;
      size_t row = (size_t)(b * S_ + kt * 64 + r);
      *(uint4v*)&Ks[r][c] = *(const uint4v*)&Kl[row * KV_W + kv * D_K + c];
      *(uint4v*)&Vs[r][c] = *(const uint4v*)&Vl[row * KV_W + kv * D_K + c];
    }
    if (kt + 1 < S_ / 64) {  // prefetch next K/V tile into cache
      for (int idx = tid; idx < 64 * 8; idx += 256) {
        int r = idx >> 3, c = (idx & 7) << 3;
        size_t row = (size_t)(b * S_ + (kt + 1) * 64 + r);
        __builtin_prefetch(&Kl[row * KV_W + kv * D_K + c], 0, 0);
        __builtin_prefetch(&Vl[row * KV_W + kv * D_K + c], 0, 0);
      }
    }
    __syncthreads();

    // S = Q @ K^T for this wave's 16 rows x 64 keys
    v8f sacc[4];
#pragma unroll
    for (int t = 0; t < 4; ++t) sacc[t] = vzero;
#pragma unroll
    for (int ch = 0; ch < 2; ++ch) {
      v16bf af = load_a_frag(&Qs[wave * 16][ch * 32], AT_LD);
#pragma unroll
      for (int nt = 0; nt < 4; ++nt) {
        v16bf bf = load_bT_frag(&Ks[nt * 16][ch * 32], AT_LD);
        sacc[nt] = wmma_bf16(af, bf, sacc[nt]);
      }
    }
#pragma unroll
    for (int nt = 0; nt < 4; ++nt)
#pragma unroll
      for (int r = 0; r < 8; ++r)
        Sf[wave * 16 + r + half * 8][nt * 16 + nc] = sacc[nt][r] * scale;
    __syncthreads();

    // per-row online softmax (128 rows, one thread each)
    if (tid < 128) {
      float mold = m_s[tid];
      float mx = mold;
      for (int j = 0; j < 64; ++j) mx = fmaxf(mx, Sf[tid][j]);
      float alpha = __expf(mold - mx);
      float sum = 0.f;
      for (int j = 0; j < 64; ++j) {
        float p = __expf(Sf[tid][j] - mx);
        sum += p;
        Ps[tid][j] = (bf16_t)p;
      }
      m_s[tid]  = mx;
      l_s[tid]  = l_s[tid] * alpha + sum;
      al_s[tid] = alpha;
    }
    __syncthreads();

    // rescale running output, then O += P @ V
#pragma unroll
    for (int r = 0; r < 8; ++r) {
      float alpha = al_s[wave * 16 + r + half * 8];
#pragma unroll
      for (int nt = 0; nt < 4; ++nt) acc[nt][r] *= alpha;
    }
#pragma unroll
    for (int ch = 0; ch < 2; ++ch) {
      v16bf af = load_a_frag(&Ps[wave * 16][ch * 32], AT_LD);
#pragma unroll
      for (int nt = 0; nt < 4; ++nt) {
        v16bf bf = load_b_frag(&Vs[ch * 32][nt * 16], AT_LD);
        acc[nt] = wmma_bf16(af, bf, acc[nt]);
      }
    }
  }
  __syncthreads();

  // normalize and store context in [b, s, h, dk] (flat [B*S, D_MODEL])
#pragma unroll
  for (int nt = 0; nt < 4; ++nt) {
#pragma unroll
    for (int r = 0; r < 8; ++r) {
      int lrow = wave * 16 + r + half * 8;
      float inv = 1.0f / l_s[lrow];
      Ctx[(size_t)(b * S_ + s0 + lrow) * D_MODEL + h * D_K + nt * 16 + nc] =
          (bf16_t)(acc[nt][r] * inv);
    }
  }
}

// ---------------------------------------------------------------------------
// host-side orchestration
// ---------------------------------------------------------------------------
extern "C" void kernel_launch(void* const* d_in, const int* in_sizes, int n_in,
                              void* d_out, int out_size, void* d_ws, size_t ws_size,
                              hipStream_t stream) {
  (void)in_sizes; (void)n_in; (void)out_size; (void)ws_size;
  const float* q  = (const float*)d_in[0];
  const float* k  = (const float*)d_in[1];
  const float* v  = (const float*)d_in[2];
  const float* Wq = (const float*)d_in[3];
  const float* bq = (const float*)d_in[4];
  const float* Wk = (const float*)d_in[5];
  const float* bk = (const float*)d_in[6];
  const float* Wv = (const float*)d_in[7];
  const float* bv = (const float*)d_in[8];
  const float* Wo = (const float*)d_in[9];
  const float* bo = (const float*)d_in[10];
  float* out = (float*)d_out;

  char* ws = (char*)d_ws;
  size_t off = 0;
  auto alloc = [&](size_t bytes) -> char* {
    char* p = ws + off;
    off += (bytes + 255) & ~(size_t)255;
    return p;
  };
  bf16_t* Xq   = (bf16_t*)alloc((size_t)MROWS * D_MODEL * 2);
  bf16_t* Xk   = (bf16_t*)alloc((size_t)MROWS * D_MODEL * 2);
  bf16_t* Xv   = (bf16_t*)alloc((size_t)MROWS * D_MODEL * 2);
  bf16_t* Wqb  = (bf16_t*)alloc((size_t)D_MODEL * D_MODEL * 2);
  bf16_t* Wkb  = (bf16_t*)alloc((size_t)KV_W * D_MODEL * 2);
  bf16_t* Wvb  = (bf16_t*)alloc((size_t)KV_W * D_MODEL * 2);
  bf16_t* Wob  = (bf16_t*)alloc((size_t)D_MODEL * D_MODEL * 2);
  bf16_t* Qlin = (bf16_t*)alloc((size_t)MROWS * D_MODEL * 2);
  bf16_t* Klin = (bf16_t*)alloc((size_t)MROWS * KV_W * 2);
  bf16_t* Vlin = (bf16_t*)alloc((size_t)MROWS * KV_W * 2);
  bf16_t* Ctx  = (bf16_t*)alloc((size_t)MROWS * D_MODEL * 2);

  auto cvt = [&](const float* src, bf16_t* dst, int n) {
    cvt_f32_bf16<<<(n + 255) / 256, 256, 0, stream>>>(src, dst, n);
  };
  cvt(q,  Xq,  MROWS * D_MODEL);
  cvt(k,  Xk,  MROWS * D_MODEL);
  cvt(v,  Xv,  MROWS * D_MODEL);
  cvt(Wq, Wqb, D_MODEL * D_MODEL);
  cvt(Wk, Wkb, KV_W * D_MODEL);
  cvt(Wv, Wvb, KV_W * D_MODEL);
  cvt(Wo, Wob, D_MODEL * D_MODEL);

  // projections (bf16 outputs)
  gemm_bf16_kernel<<<dim3(D_MODEL / 128, MROWS / 128), 256, 0, stream>>>(
      Xq, Wqb, bq, Qlin, nullptr, MROWS, D_MODEL, D_MODEL);
  gemm_bf16_kernel<<<dim3(KV_W / 128, MROWS / 128), 256, 0, stream>>>(
      Xk, Wkb, bk, Klin, nullptr, MROWS, KV_W, D_MODEL);
  gemm_bf16_kernel<<<dim3(KV_W / 128, MROWS / 128), 256, 0, stream>>>(
      Xv, Wvb, bv, Vlin, nullptr, MROWS, KV_W, D_MODEL);

  // RoPE in place on Q and K
  {
    int totq = MROWS * NHEAD * 32;
    rope_kernel<<<(totq + 255) / 256, 256, 0, stream>>>(Qlin, MROWS, NHEAD, S_);
    int totk = MROWS * NUM_KV * 32;
    rope_kernel<<<(totk + 255) / 256, 256, 0, stream>>>(Klin, MROWS, NUM_KV, S_);
  }

  // flash attention
  attn_kernel<<<dim3(S_ / 128, NHEAD, B_), 256, 0, stream>>>(Qlin, Klin, Vlin, Ctx);

  // output projection (fp32 output)
  gemm_bf16_kernel<<<dim3(D_MODEL / 128, MROWS / 128), 256, 0, stream>>>(
      Ctx, Wob, bo, nullptr, out, MROWS, D_MODEL, D_MODEL);
}
